// VAE_Decoder_Final_73572789780722
// MI455X (gfx1250) — compile-verified, matrix operand order checked by reading, not run on hardware
//
#include <hip/hip_runtime.h>
#include <hip/hip_bf16.h>

#define BATCH 4096
#define TSTEPS 32
#define ROLLN 130
#define RHYN 3
#define CONDN 12
#define Z1N 128
#define HN 1024
#define G3 3072          // 3*HN
#define KSTAT 160        // 143 static input cols padded to 160 (K%32==0)
#define NOUTP 192        // 130 output rows padded to 192 (one N block: 4 x 48-wide wave tiles)

typedef __attribute__((ext_vector_type(16))) __bf16 v16bf;
typedef __attribute__((ext_vector_type(8)))  float  v8f;

struct Frag32B { uint4 lo, hi; };

__device__ __forceinline__ unsigned short f2bf(float f) {
  unsigned int u = __float_as_uint(f);
  u = u + 0x7FFFu + ((u >> 16) & 1u);         // round-to-nearest-even
  return (unsigned short)(u >> 16);
}

// Load one 16x32 bf16 WMMA operand fragment from a K-major matrix.
// ISA layout (16-bit A/B): lanes 0-15 hold row r, K = {0..7, 16..23};
// lanes 16-31 hold row r, K = {8..15, 24..31}.  p = row base + k0.
__device__ __forceinline__ v16bf load_frag(const unsigned short* p, int h8) {
  Frag32B f;
  f.lo = *(const uint4*)(p + h8);
  f.hi = *(const uint4*)(p + 16 + h8);
  return __builtin_bit_cast(v16bf, f);
}

// C[M,N] (f32, row-major) = A[M,K] (bf16 row-major) x W[N,K]^T (bf16 row-major)
// 8 waves as 2(M) x 4(N); wave tile = (MT*16) x (NT*16).
template<int MT, int NT>
__global__ __launch_bounds__(256) void gemm_bf16_nt(
    const unsigned short* __restrict__ A,
    const unsigned short* __restrict__ W,
    float* __restrict__ C, int M, int N, int K)
{
  const int lane = threadIdx.x & 31;
  const int wave = threadIdx.x >> 5;
  const int tm = blockIdx.y * (2 * MT * 16) + (wave >> 2) * MT * 16;
  const int tn = blockIdx.x * (4 * NT * 16) + (wave & 3) * NT * 16;
  const int r  = lane & 15;
  const int h8 = (lane >> 4) << 3;

  v8f acc[MT][NT];
#pragma unroll
  for (int i = 0; i < MT; ++i)
#pragma unroll
    for (int j = 0; j < NT; ++j)
      acc[i][j] = (v8f){0.f,0.f,0.f,0.f,0.f,0.f,0.f,0.f};

  const unsigned short* Ab = A + (size_t)(tm + r) * K;
  const unsigned short* Wb = W + (size_t)(tn + r) * K;

#pragma unroll 2
  for (int k0 = 0; k0 < K; k0 += 32) {
    v16bf af[MT], wf[NT];
#pragma unroll
    for (int i = 0; i < MT; ++i) af[i] = load_frag(Ab + (size_t)i * 16 * K + k0, h8);
#pragma unroll
    for (int j = 0; j < NT; ++j) wf[j] = load_frag(Wb + (size_t)j * 16 * K + k0, h8);
#pragma unroll
    for (int i = 0; i < MT; ++i)
#pragma unroll
      for (int j = 0; j < NT; ++j)
        acc[i][j] = __builtin_amdgcn_wmma_f32_16x16x32_bf16(
            false, af[i], false, wf[j], (short)0, acc[i][j], false, false);
  }

  // C layout: element v of lane l -> row = v + 8*(l>>4), col = l&15
#pragma unroll
  for (int i = 0; i < MT; ++i) {
    const int row0 = tm + i * 16 + h8;
#pragma unroll
    for (int j = 0; j < NT; ++j) {
      float* cp = C + (size_t)row0 * N + tn + j * 16 + r;
#pragma unroll
      for (int v = 0; v < 8; ++v) cp[(size_t)v * N] = acc[i][j][v];
    }
  }
}

// GRU gate mixing.  gi/gh: [B, 3072] fp32.  emb (nullable): one-hot gather rows.
__global__ void gru_elem(const float* __restrict__ gi, const float* __restrict__ gh,
                         const float* __restrict__ emb, const int* __restrict__ amax,
                         const float* __restrict__ b_i, const float* __restrict__ b_h,
                         float* __restrict__ h, unsigned short* __restrict__ hb,
                         float* h_copy, unsigned short* hb_copy, int copyFlag)
{
  int idx = blockIdx.x * blockDim.x + threadIdx.x;
  if (idx >= BATCH * HN) return;
  int b = idx >> 10, n = idx & (HN - 1);
  const float* gib = gi + (size_t)b * G3;
  const float* ghb = gh + (size_t)b * G3;
  float er = 0.f, ez = 0.f, en = 0.f;
  if (emb) {
    const float* e = emb + (size_t)amax[b] * G3;
    er = e[n]; ez = e[n + HN]; en = e[n + 2 * HN];
  }
  float ir = gib[n]          + er + b_i[n];
  float iz = gib[n + HN]     + ez + b_i[n + HN];
  float in = gib[n + 2 * HN] + en + b_i[n + 2 * HN];
  float hr = ghb[n]          + b_h[n];
  float hz = ghb[n + HN]     + b_h[n + HN];
  float hn = ghb[n + 2 * HN] + b_h[n + 2 * HN];
  float rr = 1.f / (1.f + expf(-(ir + hr)));
  float zz = 1.f / (1.f + expf(-(iz + hz)));
  float nn = tanhf(in + rr * hn);
  float hv = (1.f - zz) * nn + zz * h[idx];
  h[idx] = hv;
  unsigned short hbv = f2bf(hv);
  hb[idx] = hbv;
  if (copyFlag) { h_copy[idx] = hv; hb_copy[idx] = hbv; }
}

// log_softmax over 130 logits + first-occurrence argmax; one block per batch row.
__global__ __launch_bounds__(128) void out_softmax(
    const float* __restrict__ logits, const float* __restrict__ b_out,
    float* __restrict__ recon, int* __restrict__ amax, int t)
{
  __shared__ float sv[128];
  __shared__ int   si[128];
  int b = blockIdx.x, tid = threadIdx.x;
  const float* lrow = logits + (size_t)b * NOUTP;
  int i1 = tid + 128;
  float x0 = (tid < ROLLN) ? lrow[tid] + b_out[tid] : -3.0e38f;
  float x1 = (i1  < ROLLN) ? lrow[i1]  + b_out[i1]  : -3.0e38f;
  float bv = x0; int bi = tid;
  if (x1 > bv) { bv = x1; bi = i1; }
  sv[tid] = bv; si[tid] = bi; __syncthreads();
  for (int s = 64; s > 0; s >>= 1) {
    if (tid < s) {
      float ov = sv[tid + s]; int oi = si[tid + s];
      if (ov > sv[tid] || (ov == sv[tid] && oi < si[tid])) { sv[tid] = ov; si[tid] = oi; }
    }
    __syncthreads();
  }
  float mx = sv[0]; int am = si[0]; __syncthreads();
  float e = 0.f;
  if (tid < ROLLN) e += expf(x0 - mx);
  if (i1  < ROLLN) e += expf(x1 - mx);
  sv[tid] = e; __syncthreads();
  for (int s = 64; s > 0; s >>= 1) { if (tid < s) sv[tid] += sv[tid + s]; __syncthreads(); }
  float lse = logf(sv[0]) + mx;
  float* out = recon + (size_t)b * TSTEPS * ROLLN + (size_t)t * ROLLN;
  if (tid < ROLLN) out[tid] = x0 - lse;
  if (i1  < ROLLN) out[i1]  = x1 - lse;
  if (tid == 0) amax[b] = am;
}

// h1_0 = tanh(z1 @ w_init^T + b_init); also seed argmax = ROLL-1 (one-hot at last).
__global__ void init_h1_kernel(const float* __restrict__ z1,
                               const float* __restrict__ w_init,
                               const float* __restrict__ b_init,
                               float* __restrict__ h1, unsigned short* __restrict__ h1b,
                               int* __restrict__ amax)
{
  int idx = blockIdx.x * blockDim.x + threadIdx.x;
  if (idx >= BATCH * HN) return;
  int b = idx >> 10, j = idx & (HN - 1);
  const float* zb = z1 + (size_t)b * Z1N;
  const float* wr = w_init + (size_t)j * Z1N;
  float s = b_init[j];
  for (int k = 0; k < Z1N; ++k) s += zb[k] * wr[k];
  float v = tanhf(s);
  h1[idx] = v; h1b[idx] = f2bf(v);
  if (j == 0) amax[b] = ROLLN - 1;
}

__global__ void cvt_bf16(const float* __restrict__ s, unsigned short* __restrict__ d, int n) {
  int i = blockIdx.x * blockDim.x + threadIdx.x;
  if (i < n) d[i] = f2bf(s[i]);
}

// w_static[n][c] = w_ih1[n][130+c] for c<143 else 0 ; [3072, 160] bf16
__global__ void build_wstatic(const float* __restrict__ w_ih1, unsigned short* __restrict__ d) {
  int i = blockIdx.x * blockDim.x + threadIdx.x;
  if (i >= G3 * KSTAT) return;
  int row = i / KSTAT, c = i - row * KSTAT;
  d[i] = (c < RHYN + Z1N + CONDN) ? f2bf(w_ih1[(size_t)row * 273 + ROLLN + c]) : (unsigned short)0;
}

// w_out_pad: [192, 1024] bf16, rows >=130 zero
__global__ void build_wout(const float* __restrict__ w_out, unsigned short* __restrict__ d) {
  int i = blockIdx.x * blockDim.x + threadIdx.x;
  if (i >= NOUTP * HN) return;
  int row = i >> 10;
  d[i] = (row < ROLLN) ? f2bf(w_out[i]) : (unsigned short)0;
}

// emb[j][n] = w_ih1[n][j], j<130 : transposed one-hot embedding rows (fp32)
__global__ void build_emb(const float* __restrict__ w_ih1, float* __restrict__ emb) {
  int i = blockIdx.x * blockDim.x + threadIdx.x;
  if (i >= ROLLN * G3) return;
  int j = i / G3, n = i - j * G3;
  emb[i] = w_ih1[(size_t)n * 273 + j];
}

// xcat[t][b][0:3]=rhythm[b][t], [3:131]=z1[b], [131:143]=cond[b][t], [143:160]=0  (bf16)
__global__ void build_xcat(const float* __restrict__ rhy, const float* __restrict__ z1,
                           const float* __restrict__ cond, unsigned short* __restrict__ x)
{
  long long i = (long long)blockIdx.x * blockDim.x + threadIdx.x;
  if (i >= (long long)TSTEPS * BATCH * KSTAT) return;
  int c = (int)(i % KSTAT);
  long long tb = i / KSTAT;
  int t = (int)(tb / BATCH), b = (int)(tb % BATCH);
  float v = 0.f;
  if (c < RHYN)                 v = rhy [(size_t)b * TSTEPS * RHYN  + (size_t)t * RHYN  + c];
  else if (c < RHYN + Z1N)      v = z1  [(size_t)b * Z1N + (c - RHYN)];
  else if (c < RHYN + Z1N + CONDN)
                                v = cond[(size_t)b * TSTEPS * CONDN + (size_t)t * CONDN + (c - RHYN - Z1N)];
  x[i] = f2bf(v);
}

__global__ void copy_f32(const float* __restrict__ s, float* __restrict__ d, int n) {
  int i = blockIdx.x * blockDim.x + threadIdx.x;
  if (i < n) d[i] = s[i];
}

extern "C" void kernel_launch(void* const* d_in, const int* in_sizes, int n_in,
                              void* d_out, int out_size, void* d_ws, size_t ws_size,
                              hipStream_t stream) {
  const float* z1      = (const float*)d_in[0];
  const float* d1m     = (const float*)d_in[1];
  const float* d1s     = (const float*)d_in[2];
  const float* d2m     = (const float*)d_in[3];
  const float* d2s     = (const float*)d_in[4];
  const float* rhy     = (const float*)d_in[5];
  const float* cond    = (const float*)d_in[7];
  const float* w_ih1   = (const float*)d_in[9];
  const float* w_hh1   = (const float*)d_in[10];
  const float* b_ih1   = (const float*)d_in[11];
  const float* b_hh1   = (const float*)d_in[12];
  const float* w_ih2   = (const float*)d_in[13];
  const float* w_hh2   = (const float*)d_in[14];
  const float* b_ih2   = (const float*)d_in[15];
  const float* b_hh2   = (const float*)d_in[16];
  const float* w_init  = (const float*)d_in[17];
  const float* b_init  = (const float*)d_in[18];
  const float* w_out   = (const float*)d_in[19];
  const float* b_out   = (const float*)d_in[20];
  float* out = (float*)d_out;

  char* wp = (char*)d_ws;
  auto alloc = [&](size_t bytes) -> void* {
    void* p = (void*)wp; wp += (bytes + 255) & ~(size_t)255; return p;
  };
  float*          bufA   = (float*)alloc((size_t)BATCH * G3 * 4);       // gi1 / gi2
  float*          bufB   = (float*)alloc((size_t)BATCH * G3 * 4);       // gh1 / gh2
  float*          logits = (float*)alloc((size_t)BATCH * NOUTP * 4);
  unsigned short* xcat   = (unsigned short*)alloc((size_t)TSTEPS * BATCH * KSTAT * 2);
  unsigned short* whh1b  = (unsigned short*)alloc((size_t)G3 * HN * 2);
  unsigned short* wih2b  = (unsigned short*)alloc((size_t)G3 * HN * 2);
  unsigned short* whh2b  = (unsigned short*)alloc((size_t)G3 * HN * 2);
  unsigned short* wstat  = (unsigned short*)alloc((size_t)G3 * KSTAT * 2);
  unsigned short* woutb  = (unsigned short*)alloc((size_t)NOUTP * HN * 2);
  float*          emb    = (float*)alloc((size_t)ROLLN * G3 * 4);
  float*          h1     = (float*)alloc((size_t)BATCH * HN * 4);
  float*          h2     = (float*)alloc((size_t)BATCH * HN * 4);
  unsigned short* h1b    = (unsigned short*)alloc((size_t)BATCH * HN * 2);
  unsigned short* h2b    = (unsigned short*)alloc((size_t)BATCH * HN * 2);
  int*            amax   = (int*)alloc((size_t)BATCH * 4);

  const int TPB = 256;
  auto blocks = [&](long long n) { return (unsigned)((n + TPB - 1) / TPB); };

  // ---- one-time prep (recomputed every call; deterministic) ----
  cvt_bf16<<<blocks(G3 * HN), TPB, 0, stream>>>(w_hh1, whh1b, G3 * HN);
  cvt_bf16<<<blocks(G3 * HN), TPB, 0, stream>>>(w_ih2, wih2b, G3 * HN);
  cvt_bf16<<<blocks(G3 * HN), TPB, 0, stream>>>(w_hh2, whh2b, G3 * HN);
  build_wstatic<<<blocks(G3 * KSTAT), TPB, 0, stream>>>(w_ih1, wstat);
  build_wout<<<blocks(NOUTP * HN), TPB, 0, stream>>>(w_out, woutb);
  build_emb<<<blocks((long long)ROLLN * G3), TPB, 0, stream>>>(w_ih1, emb);
  build_xcat<<<blocks((long long)TSTEPS * BATCH * KSTAT), TPB, 0, stream>>>(rhy, z1, cond, xcat);
  init_h1_kernel<<<blocks(BATCH * HN), TPB, 0, stream>>>(z1, w_init, b_init, h1, h1b, amax);

  // ---- passthrough outputs ----
  {
    size_t o = (size_t)BATCH * TSTEPS * ROLLN;
    copy_f32<<<blocks(BATCH * TSTEPS * RHYN), TPB, 0, stream>>>(rhy, out + o, BATCH * TSTEPS * RHYN);
    o += (size_t)BATCH * TSTEPS * RHYN;
    copy_f32<<<blocks(BATCH * Z1N), TPB, 0, stream>>>(d1m, out + o, BATCH * Z1N); o += (size_t)BATCH * Z1N;
    copy_f32<<<blocks(BATCH * Z1N), TPB, 0, stream>>>(d1s, out + o, BATCH * Z1N); o += (size_t)BATCH * Z1N;
    copy_f32<<<blocks(BATCH * Z1N), TPB, 0, stream>>>(d2m, out + o, BATCH * Z1N); o += (size_t)BATCH * Z1N;
    copy_f32<<<blocks(BATCH * Z1N), TPB, 0, stream>>>(d2s, out + o, BATCH * Z1N);
  }

  // ---- recurrent scan ----
  dim3 gMain(G3 / 128, BATCH / 128);   // 128x128 block tiles, N=3072
  dim3 gOut(1, BATCH / 128);           // 128x192 block tiles, N=192
  for (int t = 0; t < TSTEPS; ++t) {
    // gi1 (static part): xcat_t [B,160] x w_static [3072,160]^T
    gemm_bf16_nt<4, 2><<<gMain, 256, 0, stream>>>(
        xcat + (size_t)t * BATCH * KSTAT, wstat, bufA, BATCH, G3, KSTAT);
    // gh1: h1 x w_hh1^T
    gemm_bf16_nt<4, 2><<<gMain, 256, 0, stream>>>(h1b, whh1b, bufB, BATCH, G3, HN);
    // cell 1 (adds one-hot gather + biases); at t==0 copy h1 -> h2
    gru_elem<<<blocks(BATCH * HN), TPB, 0, stream>>>(
        bufA, bufB, emb, amax, b_ih1, b_hh1, h1, h1b, h2, h2b, (t == 0) ? 1 : 0);
    // gi2: h1 x w_ih2^T ; gh2: h2 x w_hh2^T
    gemm_bf16_nt<4, 2><<<gMain, 256, 0, stream>>>(h1b, wih2b, bufA, BATCH, G3, HN);
    gemm_bf16_nt<4, 2><<<gMain, 256, 0, stream>>>(h2b, whh2b, bufB, BATCH, G3, HN);
    gru_elem<<<blocks(BATCH * HN), TPB, 0, stream>>>(
        bufA, bufB, nullptr, nullptr, b_ih2, b_hh2, h2, h2b, nullptr, nullptr, 0);
    // logits: h2 x w_out_pad^T  (N padded to 192)
    gemm_bf16_nt<4, 3><<<gOut, 256, 0, stream>>>(h2b, woutb, logits, BATCH, NOUTP, HN);
    out_softmax<<<BATCH, 128, 0, stream>>>(logits, b_out, out, amax, t);
  }
}
